// PointNetEncoder_27831388078263
// MI455X (gfx1250) — matrix operand order repaired and checked by reading, not compiled, not run on hardware
//
#include <hip/hip_runtime.h>

typedef __attribute__((ext_vector_type(16))) __bf16 v16bf;
typedef __attribute__((ext_vector_type(8)))  __bf16 v8bf;
typedef __attribute__((ext_vector_type(8)))  float  v8f;

#define NN 10000
#define NE 160000

__device__ __forceinline__ float leaky_f(float v) { return v > 0.f ? v : 0.05f * v; }

// order-preserving float <-> u32 map so atomicMax(u32) == float max
__device__ __forceinline__ unsigned f2ord(float f) {
  unsigned u = __float_as_uint(f);
  return (u & 0x80000000u) ? ~u : (u | 0x80000000u);
}
__device__ __forceinline__ float ord2f(unsigned m) {
  unsigned u = (m & 0x80000000u) ? (m & 0x7FFFFFFFu) : ~m;
  return __uint_as_float(u);
}

// ---------------------------------------------------------------------------
// One-time weight swizzle: W[K x N] f32 -> fragment-ordered bf16.
// Tile (kt = k/32, nt = n/128), slot nb = (n%128)/16:
//   base = ((kt*(N/128) + nt)*8 + nb) * 512
//   elem = lane*16 + i   with lane = ((k%32)/16)*16 + (n%16),  i = k%16
// Each 32x128 B tile is 8KB *contiguous* -> LDS staging is a flat copy.
// ---------------------------------------------------------------------------
__global__ __launch_bounds__(256) void swizzleW(const float* __restrict__ W,
                                                __bf16* __restrict__ Wf,
                                                int N, int lgN, long total)
{
  long idx = (long)blockIdx.x * 256 + threadIdx.x;
  if (idx >= total) return;
  int k = (int)(idx >> lgN);
  int n = (int)(idx & (N - 1));
  int kt = k >> 5, nt = n >> 7, nb = (n >> 4) & 7;
  int lane = ((k >> 4) & 1) * 16 + (n & 15);
  int i = k & 15;
  long off = (((long)(kt * (N >> 7) + nt) * 8 + nb) * 32 + lane) * 16 + i;
  Wf[off] = (__bf16)W[idx];
}

// ---------------------------------------------------------------------------
// conv1 edge MLP: 4 -> 64 -> 128, scalar (tiny), fused atomic segment-max.
// ---------------------------------------------------------------------------
__global__ __launch_bounds__(256) void edge_mlp1(
    const float* __restrict__ x, const int* __restrict__ src, const int* __restrict__ dst,
    const float* __restrict__ w1, const float* __restrict__ b1,
    const float* __restrict__ w2, const float* __restrict__ b2,
    unsigned* __restrict__ aggU, int E)
{
  __shared__ float sW1[4 * 64];
  __shared__ float sB1[64];
  __shared__ float sW2[64 * 128];
  __shared__ float sB2[128];
  int tid = threadIdx.x;
  if (tid < 256) sW1[tid] = w1[tid];
  if (tid < 64)  sB1[tid] = b1[tid];
  for (int i = tid; i < 64 * 128; i += 256) sW2[i] = w2[i];
  if (tid < 128) sB2[tid] = b2[tid];
  __syncthreads();

  int e = blockIdx.x * 256 + tid;
  if (e >= E) return;
  int s = src[e], d = dst[e];
  float m0 = x[s * 2], m1 = x[s * 2 + 1];
  float m2 = m0 - x[d * 2], m3 = m1 - x[d * 2 + 1];

  float h[64];
#pragma unroll
  for (int c = 0; c < 64; ++c) {
    float a = sB1[c] + m0 * sW1[c] + m1 * sW1[64 + c] + m2 * sW1[128 + c] + m3 * sW1[192 + c];
    h[c] = leaky_f(a);
  }
  unsigned* base = aggU + (long)d * 128;
  for (int o = 0; o < 128; ++o) {
    float a = sB2[o];
#pragma unroll
    for (int j = 0; j < 64; ++j) a += h[j] * sW2[j * 128 + o];
    atomicMax(&base[o], f2ord(leaky_f(a)));
  }
}

// decode ordered-u32 segment-max to bf16; empty segments (decode->NaN) -> 0
__global__ __launch_bounds__(256) void finalize_agg(const unsigned* __restrict__ u,
                                                    __bf16* __restrict__ f, long n)
{
  long i = (long)blockIdx.x * 256 + threadIdx.x;
  if (i < n) {
    float v = ord2f(u[i]);
    f[i] = (__bf16)(__builtin_isfinite(v) ? v : 0.f);
  }
}

// ---------------------------------------------------------------------------
// WMMA bf16 GEMM:  C[M,N] = act(A[M,K] @ W[K,N] + bias)
//   W: fragment-ordered bf16 (pre-swizzled, zero-padded to K multiple of 32);
//      each 32x128 tile is contiguous -> flat b128 copy into LDS.
//   ASRC 0: bf16 A, K % 32 == 0      ASRC 1: conv2 edge-message gather (f32)
//   CDST 0: f32 store   CDST 1: atomicMax segment-max scatter   CDST 2: bf16
//   ACT   0: none       1: leaky     2: leaky(leaky(.))
// Block 256 thr (8 waves), tile 128x128, BK=32; A and B double-buffered in
// LDS in fragment layout. All 16 B-fragment ds_loads are issued before the
// WMMA block (scheduling fence) so waits are a staggered countdown.
// ---------------------------------------------------------------------------
#define BM 128
#define BN 128
#define BK 32

template <int ASRC, int CDST, int ACT>
__global__ __launch_bounds__(256) void wmma_gemm(
    const __bf16* __restrict__ Abf, const __bf16* __restrict__ Wf,
    const float* __restrict__ bias, float* __restrict__ C, __bf16* __restrict__ Cbf,
    int M, int K, int N,
    const float* __restrict__ hnode, const float* __restrict__ xpos,
    const int* __restrict__ src, const int* __restrict__ dst,
    unsigned* __restrict__ aggU)
{
  __shared__ __align__(32) __bf16 Asf[2][BM * BK];   // 8KB per buffer
  __shared__ __align__(32) __bf16 Bsf[2][BK * BN];   // 8KB per buffer

  const int tid  = threadIdx.x;
  const int wave = tid >> 5;       // wave32
  const int lane = tid & 31;
  const int half = lane >> 4;
  const int l16  = lane & 15;
  const int row0 = blockIdx.y * BM;
  const int col0 = blockIdx.x * BN;
  const int ktiles = (K + BK - 1) / BK;
  const int ntw = N >> 7;          // swizzled-W tile columns

  // A staging: 512 chunks of 8 consecutive k; chunk g: m=g>>2, sub=g&3
  // fragment slot: half=sub>>1, elem base=(sub&1)*8, frag_lane=half*16+(m&15)
  int a_m[2], a_k0[2], a_off[2];
#pragma unroll
  for (int j = 0; j < 2; ++j) {
    int g = j * 256 + tid;
    int m = g >> 2, sub = g & 3;
    a_m[j] = m; a_k0[j] = sub * 8;
    a_off[j] = (((m >> 4) * 32 + (sub >> 1) * 16 + (m & 15)) * 16) + ((sub & 1) * 8);
  }

  v8bf ra[2], rb[2];
  auto loadA = [&](int kt) {
#pragma unroll
    for (int j = 0; j < 2; ++j) {
      int gm = row0 + a_m[j];
      int k0 = kt * BK + a_k0[j];
      if (ASRC == 0) {                       // bf16 A, K % 32 == 0: one b128 load
        v8bf p = {};
        if (gm < M) p = *(const v8bf*)&Abf[(long)gm * K + k0];
        ra[j] = p;
      } else {                               // conv2 edge message gather (f32)
        v8f t = {};
        if (gm < M) {
          int s = src[gm];
#pragma unroll
          for (int i = 0; i < 8; ++i) {
            int gk = k0 + i;
            float v = 0.f;
            if (gk < K) {
              if (gk < K - 2) {
                v = hnode[(long)s * (K - 2) + gk];
              } else {
                int d = dst[gm];
                int c = gk - (K - 2);
                v = xpos[s * 2 + c] - xpos[d * 2 + c];
              }
            }
            t[i] = v;
          }
        }
        ra[j] = __builtin_convertvector(t, v8bf);
      }
    }
  };
  auto loadB = [&](int kt) {                 // contiguous 8KB tile, 32B/thread
    const __bf16* wb = Wf + ((long)(kt * ntw + blockIdx.x) * 8) * 512;
    rb[0] = *(const v8bf*)&wb[tid * 16];
    rb[1] = *(const v8bf*)&wb[tid * 16 + 8];
  };
  auto storeTiles = [&](int buf) {
#pragma unroll
    for (int j = 0; j < 2; ++j) *(v8bf*)&Asf[buf][a_off[j]] = ra[j];
    *(v8bf*)&Bsf[buf][tid * 16]     = rb[0];
    *(v8bf*)&Bsf[buf][tid * 16 + 8] = rb[1];
  };

  v8f acc[8] = {};   // 16 rows x 128 cols per wave

  loadA(0); loadB(0);
  storeTiles(0);
  int cur = 0;
  for (int kt = 0; kt < ktiles; ++kt) {
    const bool more = (kt + 1 < ktiles);
    if (more) { loadA(kt + 1); loadB(kt + 1); }   // global loads overlap compute
    __syncthreads();                              // LDS[cur] visible
    // issue ALL fragment loads up front, fence keeps them above the WMMAs
    v16bf afrag = *(const v16bf*)&Asf[cur][(wave * 32 + lane) * 16];
    const __bf16* bbase = &Bsf[cur][lane * 16];
    v16bf bf[8];
#pragma unroll
    for (int nb = 0; nb < 8; ++nb) bf[nb] = *(const v16bf*)&bbase[nb * 512];
    asm volatile("" ::: "memory");                // scheduling fence: no sinking
#pragma unroll
    for (int nb = 0; nb < 8; ++nb) {
      acc[nb] = __builtin_amdgcn_wmma_f32_16x16x32_bf16(
          false, afrag, false, bf[nb], (short)0, acc[nb], false, false);
    }
    if (more) storeTiles(cur ^ 1);                // other buffer: no conflict
    cur ^= 1;
  }

  // --- epilogue: bias + activation, then store / bf16 store / scatter-max ---
#pragma unroll
  for (int nb = 0; nb < 8; ++nb) {
    const int gcol = col0 + nb * 16 + l16;
    const float b = bias ? bias[gcol] : 0.f;
#pragma unroll
    for (int r = 0; r < 8; ++r) {
      const int grow = row0 + wave * 16 + (half << 3) + r;
      if (grow < M) {
        float v = acc[nb][r] + b;
        if (ACT >= 1) v = leaky_f(v);
        if (ACT == 2) v = leaky_f(v);
        if (CDST == 0)      C[(long)grow * N + gcol] = v;
        else if (CDST == 2) Cbf[(long)grow * N + gcol] = (__bf16)v;
        else atomicMax(&aggU[(long)dst[grow] * N + gcol], f2ord(v));
      }
    }
  }
}

// ---------------------------------------------------------------------------
// In-place batch norm over rows; one block per column.
// ---------------------------------------------------------------------------
__global__ __launch_bounds__(256) void batchnorm(float* __restrict__ h,
                                                 const float* __restrict__ gamma,
                                                 const float* __restrict__ beta,
                                                 int M, int N)
{
  const int col = blockIdx.x;
  const int tid = threadIdx.x;
  float s = 0.f, s2 = 0.f;
  for (int r = tid; r < M; r += 256) {
    float v = h[(long)r * N + col];
    s += v; s2 += v * v;
  }
  __shared__ float rs[256], rs2[256];
  rs[tid] = s; rs2[tid] = s2;
  __syncthreads();
  for (int off = 128; off > 0; off >>= 1) {
    if (tid < off) { rs[tid] += rs[tid + off]; rs2[tid] += rs2[tid + off]; }
    __syncthreads();
  }
  const float mean = rs[0] / (float)M;
  const float var  = rs2[0] / (float)M - mean * mean;
  const float rstd = rsqrtf(var + 1e-5f);
  const float g = gamma[col], bb = beta[col];
  for (int r = tid; r < M; r += 256) {
    long i = (long)r * N + col;
    h[i] = g * (h[i] - mean) * rstd + bb;
  }
}

// ---------------------------------------------------------------------------
extern "C" void kernel_launch(void* const* d_in, const int* in_sizes, int n_in,
                              void* d_out, int out_size, void* d_ws, size_t ws_size,
                              hipStream_t stream) {
  (void)in_sizes; (void)n_in; (void)out_size; (void)ws_size;
  const float* x    = (const float*)d_in[0];
  const int*   ei   = (const int*)d_in[1];
  const int* src = ei;
  const int* dst = ei + NE;
  const float* l1w1 = (const float*)d_in[3];
  const float* l1b1 = (const float*)d_in[4];
  const float* l1w2 = (const float*)d_in[5];
  const float* l1b2 = (const float*)d_in[6];
  const float* g1w1 = (const float*)d_in[7];
  const float* g1b1 = (const float*)d_in[8];
  const float* g1w2 = (const float*)d_in[9];
  const float* g1b2 = (const float*)d_in[10];
  const float* gamma1 = (const float*)d_in[11];
  const float* beta1  = (const float*)d_in[12];
  const float* l2w1 = (const float*)d_in[13];
  const float* l2b1 = (const float*)d_in[14];
  const float* l2w2 = (const float*)d_in[15];
  const float* l2b2 = (const float*)d_in[16];
  const float* g2w1 = (const float*)d_in[17];
  const float* g2b1 = (const float*)d_in[18];
  const float* g2w2 = (const float*)d_in[19];
  const float* g2b2 = (const float*)d_in[20];
  const float* gamma2 = (const float*)d_in[21];
  const float* beta2  = (const float*)d_in[22];
  float* out = (float*)d_out;

  // workspace carve-out (~480 MB)
  char* wsb = (char*)d_ws;
  size_t off = 0;
  auto take = [&](size_t bytes) { char* p = wsb + off; off += (bytes + 255) & ~(size_t)255; return p; };
  unsigned* agg1U  = (unsigned*)take((size_t)NN * 128 * 4);
  __bf16*   agg1b  = (__bf16*)  take((size_t)NN * 128 * 2);
  __bf16*   t1b    = (__bf16*)  take((size_t)NN * 256 * 2);
  float*    hnode  = (float*)   take((size_t)NN * 512 * 4);
  __bf16*   h2ab   = (__bf16*)  take((size_t)NE * 1024 * 2);
  unsigned* agg2U  = (unsigned*)take((size_t)NN * 1024 * 4);
  __bf16*   agg2b  = (__bf16*)  take((size_t)NN * 1024 * 2);
  __bf16*   n2ab   = (__bf16*)  take((size_t)NN * 2048 * 2);
  // fragment-ordered bf16 weights (K padded up to multiple of 32)
  __bf16* g1w1f = (__bf16*)take((size_t)128  * 256  * 2);
  __bf16* g1w2f = (__bf16*)take((size_t)256  * 512  * 2);
  __bf16* l2w1f = (__bf16*)take((size_t)544  * 1024 * 2);   // K=514 -> pad 544
  __bf16* l2w2f = (__bf16*)take((size_t)1024 * 1024 * 2);
  __bf16* g2w1f = (__bf16*)take((size_t)1024 * 2048 * 2);
  __bf16* g2w2f = (__bf16*)take((size_t)2048 * 2048 * 2);

  // one-time weight swizzles (L2-resident for the whole run)
  hipMemsetAsync(l2w1f, 0, (size_t)544 * 1024 * 2, stream);   // zero K-tail pad
  auto swz = [&](const float* w, __bf16* wf, int K, int N, int lgN) {
    long total = (long)K * N;
    swizzleW<<<(unsigned)((total + 255) / 256), 256, 0, stream>>>(w, wf, N, lgN, total);
  };
  swz(g1w1, g1w1f, 128,  256,  8);
  swz(g1w2, g1w2f, 256,  512,  9);
  swz(l2w1, l2w1f, 514,  1024, 10);
  swz(l2w2, l2w2f, 1024, 1024, 10);
  swz(g2w1, g2w1f, 1024, 2048, 11);
  swz(g2w2, g2w2f, 2048, 2048, 11);

  // ordered-u32 init: 0 decodes to NaN -> finalize maps empty segments to 0
  hipMemsetAsync(agg1U, 0, (size_t)NN * 128 * 4, stream);
  hipMemsetAsync(agg2U, 0, (size_t)NN * 1024 * 4, stream);

  const int gy = (NN + BM - 1) / BM;   // 79
  // --- conv1 ---
  edge_mlp1<<<(NE + 255) / 256, 256, 0, stream>>>(x, src, dst, l1w1, l1b1, l1w2, l1b2,
                                                  agg1U, NE);
  finalize_agg<<<(NN * 128 + 255) / 256, 256, 0, stream>>>(agg1U, agg1b, (long)NN * 128);
  wmma_gemm<0, 2, 1><<<dim3(256 / BN, gy), 256, 0, stream>>>(
      agg1b, g1w1f, g1b1, nullptr, t1b, NN, 128, 256,
      nullptr, nullptr, nullptr, nullptr, nullptr);
  wmma_gemm<0, 0, 2><<<dim3(512 / BN, gy), 256, 0, stream>>>(
      t1b, g1w2f, g1b2, hnode, nullptr, NN, 256, 512,
      nullptr, nullptr, nullptr, nullptr, nullptr);
  batchnorm<<<512, 256, 0, stream>>>(hnode, gamma1, beta1, NN, 512);

  // --- conv2: gather fused into A-stage, segment-max fused into epilogue ---
  wmma_gemm<1, 2, 1><<<dim3(1024 / BN, NE / BM), 256, 0, stream>>>(
      nullptr, l2w1f, l2b1, nullptr, h2ab, NE, 514, 1024,
      hnode, x, src, dst, nullptr);
  wmma_gemm<0, 1, 1><<<dim3(1024 / BN, NE / BM), 256, 0, stream>>>(
      h2ab, l2w2f, l2b2, nullptr, nullptr, NE, 1024, 1024,
      nullptr, nullptr, nullptr, dst, agg2U);
  finalize_agg<<<(NN * 1024 + 255) / 256, 256, 0, stream>>>(agg2U, agg2b, (long)NN * 1024);
  wmma_gemm<0, 2, 1><<<dim3(2048 / BN, gy), 256, 0, stream>>>(
      agg2b, g2w1f, g2b1, nullptr, n2ab, NN, 1024, 2048,
      nullptr, nullptr, nullptr, nullptr, nullptr);
  wmma_gemm<0, 0, 2><<<dim3(2048 / BN, gy), 256, 0, stream>>>(
      n2ab, g2w2f, g2b2, out, nullptr, NN, 2048, 2048,
      nullptr, nullptr, nullptr, nullptr, nullptr);
  batchnorm<<<2048, 256, 0, stream>>>(out, gamma2, beta2, NN, 2048);
}
